// SlotPixelPartGraphMotif_27315992002774
// MI455X (gfx1250) — matrix operand order repaired and checked by reading, not compiled, not run on hardware
//
#include <hip/hip_runtime.h>
#include <math.h>

// ---------------------------------------------------------------------------
// Problem constants (from the reference)
// ---------------------------------------------------------------------------
#define Bq   8
#define Nn_  9216
#define HDq  128
#define Eq   73728
#define Kq   16
#define Cq   7
#define BNq  (Bq * Nn_)      // 73728
#define BEq  ((long long)Bq * Eq) // 589824

typedef _Float16 half_t;
typedef __attribute__((ext_vector_type(16))) _Float16 v16h;
typedef __attribute__((ext_vector_type(8)))  _Float16 v8h;
typedef __attribute__((ext_vector_type(8)))  float    v8f;
typedef __attribute__((ext_vector_type(4)))  float    v4f;

__device__ __forceinline__ float gelu_f(float x) {
    return 0.5f * x * (1.0f + erff(x * 0.70710678118654752f));
}

// ---------------------------------------------------------------------------
// WMMA fragment loaders (layouts per CDNA5 ISA §7.12.2)
// A (16x32 f16): lane holds row (lane&15), K = {koff..koff+7, koff+16..koff+23},
//                koff = (lane>>4)*8.  Source is fp32 -> convert in registers.
// B (32x16 f16): lane holds col (lane&15), K = {koff..koff+7, koff+16..koff+23}, koff = (lane>>4)*8.
//                Weights pre-stored transposed: WT[n][k], so B loads are
//                two contiguous 16B v8h loads.
// ---------------------------------------------------------------------------
__device__ __forceinline__ v16h load_a_f32(const float* __restrict__ p) {
    v4f a0 = *(const v4f*)(p);
    v4f a1 = *(const v4f*)(p + 4);
    v4f a2 = *(const v4f*)(p + 16);
    v4f a3 = *(const v4f*)(p + 20);
    v16h r;
#pragma unroll
    for (int i = 0; i < 4; ++i) {
        r[i]      = (_Float16)a0[i];
        r[4 + i]  = (_Float16)a1[i];
        r[8 + i]  = (_Float16)a2[i];
        r[12 + i] = (_Float16)a3[i];
    }
    return r;
}

__device__ __forceinline__ v16h load_b_f16(const half_t* __restrict__ p) {
    v8h b0 = *(const v8h*)(p);
    v8h b1 = *(const v8h*)(p + 16);
    v16h r;
#pragma unroll
    for (int i = 0; i < 8; ++i) { r[i] = b0[i]; r[8 + i] = b1[i]; }
    return r;
}

// ---------------------------------------------------------------------------
// Generic fused linear:  Out[z] = act(alpha * A[z] @ WT^T + bias) (/ divRow)
// One wave (32 threads) per block; wave computes a 16 x (16*NT) output tile
// via v_wmma_f32_16x16x32_f16.  grid = (M/16, Nout/(16*NT), Z).
// ACT: 0 = none, 1 = GELU, 2 = sigmoid.
// ---------------------------------------------------------------------------
template <int ACT, int NT>
__global__ __launch_bounds__(32) void gemm_lin(
    const float* __restrict__ A, long long sAz,
    const half_t* __restrict__ WT, long long sWz,
    const float* __restrict__ bias,
    float* __restrict__ Out, long long sOz,
    int K, int Nout, float alpha,
    const float* __restrict__ divRow, int divStride, int accumulate)
{
    const int lane  = threadIdx.x;
    const int r     = lane & 15;
    const int koffA = (lane >> 4) * 8;
    const int koffB = (lane >> 4) * 8;
    const long long row0 = (long long)blockIdx.x * 16;
    const int col0 = blockIdx.y * (16 * NT);

    const float*  Ab = A  + (long long)blockIdx.z * sAz + (row0 + r) * (long long)K + koffA;
    const half_t* Wb = WT + (long long)blockIdx.z * sWz;
    float*        Ob = Out + (long long)blockIdx.z * sOz;

    v8f acc[NT] = {};
    for (int k0 = 0; k0 < K; k0 += 32) {
        v16h a = load_a_f32(Ab + k0);
#pragma unroll
        for (int t = 0; t < NT; ++t) {
            const half_t* wp = Wb + (long long)(col0 + t * 16 + r) * K + k0 + koffB;
            v16h b = load_b_f16(wp);
            acc[t] = __builtin_amdgcn_wmma_f32_16x16x32_f16(
                false, a, false, b, (short)0, acc[t], false, false);
        }
        if (k0 + 64 <= K) __builtin_prefetch(Ab + k0 + 64, 0, 1); // global_prefetch
    }

    const int n_ = lane & 15;
    const int mo = (lane >> 4) * 8;
#pragma unroll
    for (int v = 0; v < 8; ++v) {
        const long long m = row0 + mo + v;
        float dv = 1.0f;
        if (divRow) dv = 1.0f / fmaxf(divRow[(long long)blockIdx.z * divStride + m], 1e-6f);
        float* orow = Ob + m * (long long)Nout;
#pragma unroll
        for (int t = 0; t < NT; ++t) {
            const int c = col0 + t * 16 + n_;
            float val = acc[t][v] * alpha + (bias ? bias[c] : 0.0f);
            if (ACT == 1)      val = gelu_f(val);
            else if (ACT == 2) val = 1.0f / (1.0f + expf(-val));
            val *= dv;
            if (accumulate) orow[c] += val; else orow[c] = val;
        }
    }
}

// ---------------------------------------------------------------------------
// Fused edge-message GEMM: for edge rows (b,e) compute
//   msg = gelu(h[b, src[e]] @ msgW + bias) * gate[e]
// and scatter-add into agg[b, dst[e]] with global_atomic_add_f32.
// grid = (B*E/16, 2); wave tile 16 rows x 64 cols; K = 128 fixed.
// ---------------------------------------------------------------------------
__global__ __launch_bounds__(32) void msg_gemm(
    const float* __restrict__ h, const half_t* __restrict__ WT,
    const float* __restrict__ bias,
    const int* __restrict__ srcI, const int* __restrict__ dstI,
    const float* __restrict__ gate,
    float* __restrict__ agg, int Nn, int Ee)
{
    const int lane  = threadIdx.x;
    const int r     = lane & 15;
    const int koffA = (lane >> 4) * 8;
    const int koffB = (lane >> 4) * 8;
    const long long rb = (long long)blockIdx.x * 16;
    const int b     = (int)(rb / Ee);
    const int ebase = (int)(rb % Ee);
    const int col0  = blockIdx.y * 64;

    const int sr = srcI[ebase + r];                       // row gather
    const float* Ab = h + ((long long)b * Nn + sr) * HDq + koffA;

    v8f acc[4] = {};
    for (int k0 = 0; k0 < HDq; k0 += 32) {
        v16h a = load_a_f32(Ab + k0);
#pragma unroll
        for (int t = 0; t < 4; ++t) {
            v16h bb = load_b_f16(WT + (long long)(col0 + t * 16 + r) * HDq + k0 + koffB);
            acc[t] = __builtin_amdgcn_wmma_f32_16x16x32_f16(
                false, a, false, bb, (short)0, acc[t], false, false);
        }
    }

    const int n_ = lane & 15;
    const int mo = (lane >> 4) * 8;
#pragma unroll
    for (int v = 0; v < 8; ++v) {
        const int e = ebase + mo + v;
        const int d = dstI[e];
        const float* gr = gate + (long long)e * HDq;
        float* orow = agg + ((long long)b * Nn + d) * HDq;
#pragma unroll
        for (int t = 0; t < 4; ++t) {
            const int c = col0 + t * 16 + n_;
            float val = gelu_f(acc[t][v] + bias[c]) * gr[c];
            atomicAdd(orow + c, val);                     // global_atomic_add_f32
        }
    }
}

// ---------------------------------------------------------------------------
// Weight prep: WT16[n*K+k] = (f16) W[k*N+n]   (optionally batched via grid.z)
// ---------------------------------------------------------------------------
__global__ void prep_wt(const float* __restrict__ W, half_t* __restrict__ WT,
                        int K, int N, long long sWz, long long sTz)
{
    const float* w = W + (long long)blockIdx.z * sWz;
    half_t* t      = WT + (long long)blockIdx.z * sTz;
    long long idx = (long long)blockIdx.x * blockDim.x + threadIdx.x;
    if (idx >= (long long)K * N) return;
    int k = (int)(idx / N), n = (int)(idx % N);
    t[(long long)n * K + k] = (half_t)w[idx];
}

// normalize part queries and store as f16 WT (already [n=part][k=feat] layout)
__global__ __launch_bounds__(32) void qnorm_k(const float* __restrict__ q,
                                              half_t* __restrict__ q16)
{
    const int kk = blockIdx.x, lane = threadIdx.x;
    float v[4]; float ss = 0.f;
#pragma unroll
    for (int i = 0; i < 4; ++i) { v[i] = q[kk * HDq + lane + 32 * i]; ss += v[i] * v[i]; }
    for (int o = 16; o; o >>= 1) ss += __shfl_xor(ss, o, 32);
    const float nrm = fmaxf(sqrtf(ss), 1e-6f);
#pragma unroll
    for (int i = 0; i < 4; ++i) q16[kk * HDq + lane + 32 * i] = (half_t)(v[i] / nrm);
}

// ---------------------------------------------------------------------------
// Input projection + LayerNorm + GELU + node mask (K = 7, one wave per row)
// ---------------------------------------------------------------------------
__global__ __launch_bounds__(32) void input_proj(
    const float* __restrict__ x, const float* __restrict__ W,
    const float* __restrict__ bias, const float* __restrict__ g,
    const float* __restrict__ bln, const unsigned char* __restrict__ mask,
    float* __restrict__ h)
{
    const long long row = blockIdx.x;
    const int lane = threadIdx.x;
    float xv[7];
#pragma unroll
    for (int k = 0; k < 7; ++k) xv[k] = x[row * 7 + k];
    float s[4]; float mean = 0.f;
#pragma unroll
    for (int i = 0; i < 4; ++i) {
        const int c = lane + 32 * i;
        float a = bias[c];
#pragma unroll
        for (int k = 0; k < 7; ++k) a += xv[k] * W[k * HDq + c];
        s[i] = a; mean += a;
    }
    for (int o = 16; o; o >>= 1) mean += __shfl_xor(mean, o, 32);
    mean *= (1.0f / HDq);
    float var = 0.f;
#pragma unroll
    for (int i = 0; i < 4; ++i) { float d = s[i] - mean; var += d * d; }
    for (int o = 16; o; o >>= 1) var += __shfl_xor(var, o, 32);
    var *= (1.0f / HDq);
    const float rstd = rsqrtf(var + 1e-5f);
    const float fm = (float)mask[row];
#pragma unroll
    for (int i = 0; i < 4; ++i) {
        const int c = lane + 32 * i;
        h[row * HDq + c] = gelu_f((s[i] - mean) * rstd * g[c] + bln[c]) * fm;
    }
}

// edge gate stage 1:  g1 = gelu(edge_attr @ eg1 + b)   (K = 5)
__global__ __launch_bounds__(32) void edge_g1(
    const float* __restrict__ ea, const float* __restrict__ W,
    const float* __restrict__ bias, float* __restrict__ out)
{
    const long long row = blockIdx.x;
    const int lane = threadIdx.x;
    float ev[5];
#pragma unroll
    for (int k = 0; k < 5; ++k) ev[k] = ea[row * 5 + k];
#pragma unroll
    for (int i = 0; i < 4; ++i) {
        const int c = lane + 32 * i;
        float a = bias[c];
#pragma unroll
        for (int k = 0; k < 5; ++k) a += ev[k] * W[k * HDq + c];
        out[row * HDq + c] = gelu_f(a);
    }
}

// Out = LayerNorm(X + T)  (D = 128, one wave per row; Out may alias X)
__global__ __launch_bounds__(32) void ln_res(
    const float* __restrict__ X, const float* __restrict__ T,
    const float* __restrict__ g, const float* __restrict__ b,
    float* __restrict__ Out)
{
    const long long row = blockIdx.x;
    const int lane = threadIdx.x;
    float s[4]; float mean = 0.f;
#pragma unroll
    for (int i = 0; i < 4; ++i) {
        const long long idx = row * HDq + lane + 32 * i;
        s[i] = X[idx] + T[idx]; mean += s[i];
    }
    for (int o = 16; o; o >>= 1) mean += __shfl_xor(mean, o, 32);
    mean *= (1.0f / HDq);
    float var = 0.f;
#pragma unroll
    for (int i = 0; i < 4; ++i) { float d = s[i] - mean; var += d * d; }
    for (int o = 16; o; o >>= 1) var += __shfl_xor(var, o, 32);
    var *= (1.0f / HDq);
    const float rstd = rsqrtf(var + 1e-5f);
#pragma unroll
    for (int i = 0; i < 4; ++i) {
        const int c = lane + 32 * i;
        Out[row * HDq + c] = (s[i] - mean) * rstd * g[c] + b[c];
    }
}

__global__ void zero_f(float* __restrict__ p, long long n) {
    long long i = (long long)blockIdx.x * blockDim.x + threadIdx.x;
    if (i < n) p[i] = 0.0f;
}

__global__ void deg_k(const int* __restrict__ dst, float* __restrict__ deg, int Ee) {
    int i = blockIdx.x * blockDim.x + threadIdx.x;
    if (i < Ee) atomicAdd(&deg[dst[i]], 1.0f);
}

__global__ void scale_agg(float* __restrict__ agg, const float* __restrict__ deg,
                          const unsigned char* __restrict__ mask, int Nn, long long total)
{
    long long idx = (long long)blockIdx.x * blockDim.x + threadIdx.x;
    if (idx >= total) return;
    long long row = idx >> 7;
    int n = (int)(row % Nn);
    agg[idx] = agg[idx] / fmaxf(deg[n], 1.0f) * (float)mask[row];
}

__global__ void mask_mul(float* __restrict__ h, const unsigned char* __restrict__ mask,
                         long long total)
{
    long long idx = (long long)blockIdx.x * blockDim.x + threadIdx.x;
    if (idx >= total) return;
    h[idx] *= (float)mask[idx >> 7];
}

// softmax over K=16 parts per (b,n); emits pm, pmT, and column sums (atomics)
__global__ void softmax_pool(const float* __restrict__ logits,
                             const unsigned char* __restrict__ mask,
                             float* __restrict__ pm, float* __restrict__ pmT,
                             float* __restrict__ sums, int Nn)
{
    long long row = (long long)blockIdx.x * blockDim.x + threadIdx.x;
    if (row >= BNq) return;
    const float fm = (float)mask[row];
    float l[Kq]; float mx = -1e30f;
#pragma unroll
    for (int k = 0; k < Kq; ++k) {
        l[k] = (fm != 0.0f) ? logits[row * Kq + k] : -10000.0f;
        mx = fmaxf(mx, l[k]);
    }
    float se = 0.f;
#pragma unroll
    for (int k = 0; k < Kq; ++k) { l[k] = expf(l[k] - mx); se += l[k]; }
    const int b = (int)(row / Nn), n = (int)(row % Nn);
#pragma unroll
    for (int k = 0; k < Kq; ++k) {
        const float p = l[k] / se * fm;
        pm[row * Kq + k] = p;
        pmT[((long long)(b * Kq + k)) * Nn + n] = p;
        atomicAdd(&sums[b * Kq + k], p);
    }
}

// centers[b,k] = sum_n pool * pos(n);  pos derived analytically from n
__global__ __launch_bounds__(32) void centers_k(const float* __restrict__ pmT,
                                                const float* __restrict__ sums,
                                                float* __restrict__ centers, int Nn)
{
    const int bk = blockIdx.x, lane = threadIdx.x;
    const float* p = pmT + (long long)bk * Nn;
    float cx = 0.f, cy = 0.f;
    const float inv95 = 1.0f / 95.0f;
    for (int n = lane; n < Nn; n += 32) {
        const float w = p[n];
        cx += w * (float)(n % 96) * inv95;
        cy += w * (float)(n / 96) * inv95;
    }
    for (int o = 16; o; o >>= 1) { cx += __shfl_xor(cx, o, 32); cy += __shfl_xor(cy, o, 32); }
    if (lane == 0) {
        const float s = fmaxf(sums[bk], 1e-6f);
        centers[bk * 2 + 0] = cx / s;
        centers[bk * 2 + 1] = cy / s;
    }
}

// t = gelu(centers @ pos1.w + b)   (K = 2)
__global__ __launch_bounds__(128) void pos1_k(const float* __restrict__ centers,
                                              const float* __restrict__ W,
                                              const float* __restrict__ bias,
                                              float* __restrict__ out)
{
    const int row = blockIdx.x, f = threadIdx.x;
    out[row * HDq + f] = gelu_f(centers[row * 2] * W[f] +
                                centers[row * 2 + 1] * W[HDq + f] + bias[f]);
}

// tiny 4-head self attention over K=16 tokens: grid = B*4 blocks of 32
__global__ __launch_bounds__(32) void part_attn(const float* __restrict__ qkv,
                                                float* __restrict__ ao)
{
    const int bh = blockIdx.x, b = bh >> 2, hseg = bh & 3;
    const int lane = threadIdx.x;
    if (lane >= Kq) return;
    const float* qr = qkv + (long long)(b * Kq + lane) * 384 + hseg * 32;
    float lg[Kq]; float mx = -1e30f;
    const float sc = 0.17677669529663687f; // 1/sqrt(32)
#pragma unroll
    for (int j = 0; j < Kq; ++j) {
        const float* kr = qkv + (long long)(b * Kq + j) * 384 + 128 + hseg * 32;
        float d = 0.f;
#pragma unroll
        for (int c = 0; c < 32; ++c) d += qr[c] * kr[c];
        lg[j] = d * sc; mx = fmaxf(mx, lg[j]);
    }
    float se = 0.f;
#pragma unroll
    for (int j = 0; j < Kq; ++j) { lg[j] = expf(lg[j] - mx); se += lg[j]; }
    float o[32] = {};
#pragma unroll
    for (int j = 0; j < Kq; ++j) {
        const float* vr = qkv + (long long)(b * Kq + j) * 384 + 256 + hseg * 32;
        const float w = lg[j] / se;
#pragma unroll
        for (int c = 0; c < 32; ++c) o[c] += w * vr[c];
    }
#pragma unroll
    for (int c = 0; c < 32; ++c)
        ao[(long long)(b * Kq + lane) * HDq + hseg * 32 + c] = o[c];
}

// mean over K, then cls1 (gelu) -> cls2 head, via LDS
__global__ __launch_bounds__(128) void head_k(const float* __restrict__ pc,
                                              const float* __restrict__ c1w,
                                              const float* __restrict__ c1b,
                                              const float* __restrict__ c2w,
                                              const float* __restrict__ c2b,
                                              float* __restrict__ out)
{
    __shared__ float feat[HDq];
    __shared__ float h1[HDq];
    const int b = blockIdx.x, f = threadIdx.x;
    float a = 0.f;
#pragma unroll
    for (int k = 0; k < Kq; ++k) a += pc[(long long)(b * Kq + k) * HDq + f];
    feat[f] = a * (1.0f / Kq);
    __syncthreads();
    float s = c1b[f];
    for (int k = 0; k < HDq; ++k) s += feat[k] * c1w[k * HDq + f];
    h1[f] = gelu_f(s);
    __syncthreads();
    if (f < Cq) {
        float o = c2b[f];
        for (int k = 0; k < HDq; ++k) o += h1[k] * c2w[k * Cq + f];
        out[b * Cq + f] = o;
    }
}

// ---------------------------------------------------------------------------
// Host launcher
// ---------------------------------------------------------------------------
struct LinP { const float* b; const float* w; };
struct LnP  { const float* b; const float* g; };

extern "C" void kernel_launch(void* const* d_in, const int* in_sizes, int n_in,
                              void* d_out, int out_size, void* d_ws, size_t ws_size,
                              hipStream_t stream)
{
    (void)in_sizes; (void)n_in; (void)out_size; (void)ws_size;

    // ---- inputs (setup_inputs insertion order, params leaves jax-tree order)
    const float*         x     = (const float*)d_in[0];
    const int*           ei    = (const int*)d_in[1];
    const int*           srcI  = ei;
    const int*           dstI  = ei + Eq;
    const float*         eattr = (const float*)d_in[2];
    const unsigned char* maskp = (const unsigned char*)d_in[3];

    auto F = [&](int i) { return (const float*)d_in[i]; };
    LinP cls1   = {F(4),  F(5)};
    LinP cls2   = {F(6),  F(7)};
    LnP  in_ln  = {F(8),  F(9)};
    LinP in_pj  = {F(10), F(11)};
    LinP p_ffn1 = {F(12), F(13)};
    LinP p_ffn2 = {F(14), F(15)};
    LinP p_inp  = {F(16), F(17)};
    LnP  p_ln1  = {F(18), F(19)};
    LnP  p_ln2  = {F(20), F(21)};
    LinP p_out  = {F(22), F(23)};
    const float* part_q = F(24);
    LinP pix_key = {F(25), F(26)};
    LinP agg_l[2], eg1_l[2], eg2_l[2], ffn1_l[2], ffn2_l[2], msg_l[2];
    LnP  ln1_l[2], ln2_l[2];
    for (int l = 0; l < 2; ++l) {
        const int b0 = 27 + l * 16;
        agg_l[l]  = {F(b0 + 0),  F(b0 + 1)};
        eg1_l[l]  = {F(b0 + 2),  F(b0 + 3)};
        eg2_l[l]  = {F(b0 + 4),  F(b0 + 5)};
        ffn1_l[l] = {F(b0 + 6),  F(b0 + 7)};
        ffn2_l[l] = {F(b0 + 8),  F(b0 + 9)};
        ln1_l[l]  = {F(b0 + 10), F(b0 + 11)};
        ln2_l[l]  = {F(b0 + 12), F(b0 + 13)};
        msg_l[l]  = {F(b0 + 14), F(b0 + 15)};
    }
    LinP pix_val = {F(59), F(60)};
    LinP pos1    = {F(61), F(62)};
    LinP pos2    = {F(63), F(64)};

    // ---- workspace carve (256B aligned)
    char* wp = (char*)d_ws;
    auto carve = [&](size_t bytes) -> void* {
        void* p = (void*)wp;
        wp += (bytes + 255) & ~(size_t)255;
        return p;
    };
    float* h    = (float*)carve((size_t)BNq * HDq * 4);
    float* t256 = (float*)carve((size_t)BNq * 256 * 4);
    float* t128 = (float*)carve((size_t)BNq * HDq * 4);
    float* agg  = (float*)carve((size_t)BNq * HDq * 4);
    float* gate = (float*)carve((size_t)BNq * HDq * 4);
    float* deg  = (float*)carve((size_t)Nn_ * 4);
    float* pm   = (float*)carve((size_t)BNq * Kq * 4);
    float* pmT  = (float*)carve((size_t)Bq * Kq * Nn_ * 4);
    float* sums = (float*)carve(128 * 4);
    half_t* vT16 = (half_t*)carve((size_t)Bq * HDq * Nn_ * 2);
    float* pf   = (float*)carve(128 * HDq * 4);
    float* cen  = (float*)carve(128 * 2 * 4);
    float* pqkv = (float*)carve(128 * 384 * 4);
    float* ao   = (float*)carve(128 * HDq * 4);
    float* pt   = (float*)carve(128 * HDq * 4);
    float* p256 = (float*)carve(128 * 256 * 4);
    half_t* w_eg2[2], *w_msg[2], *w_agg[2], *w_ffn1[2], *w_ffn2[2];
    for (int l = 0; l < 2; ++l) {
        w_eg2[l]  = (half_t*)carve(128 * 128 * 2);
        w_msg[l]  = (half_t*)carve(128 * 128 * 2);
        w_agg[l]  = (half_t*)carve(128 * 128 * 2);
        w_ffn1[l] = (half_t*)carve(128 * 256 * 2);
        w_ffn2[l] = (half_t*)carve(256 * 128 * 2);
    }
    half_t* w_key   = (half_t*)carve(128 * 128 * 2);
    half_t* w_val   = (half_t*)carve(128 * 128 * 2);
    half_t* w_q     = (half_t*)carve(16 * 128 * 2);
    half_t* w_pin   = (half_t*)carve(128 * 384 * 2);
    half_t* w_pout  = (half_t*)carve(128 * 128 * 2);
    half_t* w_pffn1 = (half_t*)carve(128 * 256 * 2);
    half_t* w_pffn2 = (half_t*)carve(256 * 128 * 2);
    half_t* w_pos2  = (half_t*)carve(128 * 128 * 2);

    const long long tot = (long long)BNq * HDq;
    auto eg = [](long long n) { return (unsigned)((n + 255) / 256); };

    // ---- weight prep (f16 transposed) + degree
    auto prep = [&](const float* w, half_t* wt, int K, int N) {
        prep_wt<<<dim3(eg((long long)K * N), 1, 1), 256, 0, stream>>>(w, wt, K, N, 0, 0);
    };
    for (int l = 0; l < 2; ++l) {
        prep(eg2_l[l].w,  w_eg2[l],  128, 128);
        prep(msg_l[l].w,  w_msg[l],  128, 128);
        prep(agg_l[l].w,  w_agg[l],  128, 128);
        prep(ffn1_l[l].w, w_ffn1[l], 128, 256);
        prep(ffn2_l[l].w, w_ffn2[l], 256, 128);
    }
    prep(pix_key.w, w_key, 128, 128);
    prep(pix_val.w, w_val, 128, 128);
    prep(p_inp.w,  w_pin,   128, 384);
    prep(p_out.w,  w_pout,  128, 128);
    prep(p_ffn1.w, w_pffn1, 128, 256);
    prep(p_ffn2.w, w_pffn2, 256, 128);
    prep(pos2.w,   w_pos2,  128, 128);
    qnorm_k<<<16, 32, 0, stream>>>(part_q, w_q);

    zero_f<<<eg(Nn_), 256, 0, stream>>>(deg, Nn_);
    deg_k<<<eg(Eq), 256, 0, stream>>>(dstI, deg, Eq);

    // ---- input projection
    input_proj<<<BNq, 32, 0, stream>>>(x, in_pj.w, in_pj.b, in_ln.g, in_ln.b, maskp, h);

    // ---- pixel (message passing) layers
    for (int l = 0; l < 2; ++l) {
        edge_g1<<<Eq, 32, 0, stream>>>(eattr, eg1_l[l].w, eg1_l[l].b, t128);
        gemm_lin<2, 4><<<dim3(Eq / 16, 2, 1), 32, 0, stream>>>(
            t128, 0, w_eg2[l], 0, eg2_l[l].b, gate, 0, 128, 128, 1.f, nullptr, 0, 0);
        zero_f<<<eg(tot), 256, 0, stream>>>(agg, tot);
        msg_gemm<<<dim3((unsigned)(BEq / 16), 2, 1), 32, 0, stream>>>(
            h, w_msg[l], msg_l[l].b, srcI, dstI, gate, agg, Nn_, Eq);
        scale_agg<<<eg(tot), 256, 0, stream>>>(agg, deg, maskp, Nn_, tot);
        gemm_lin<1, 4><<<dim3(BNq / 16, 2, 1), 32, 0, stream>>>(
            agg, 0, w_agg[l], 0, agg_l[l].b, t128, 0, 128, 128, 1.f, nullptr, 0, 0);
        ln_res<<<BNq, 32, 0, stream>>>(h, t128, ln1_l[l].g, ln1_l[l].b, h);
        gemm_lin<1, 4><<<dim3(BNq / 16, 4, 1), 32, 0, stream>>>(
            h, 0, w_ffn1[l], 0, ffn1_l[l].b, t256, 0, 128, 256, 1.f, nullptr, 0, 0);
        gemm_lin<0, 4><<<dim3(BNq / 16, 2, 1), 32, 0, stream>>>(
            t256, 0, w_ffn2[l], 0, ffn2_l[l].b, t128, 0, 256, 128, 1.f, nullptr, 0, 0);
        ln_res<<<BNq, 32, 0, stream>>>(h, t128, ln2_l[l].g, ln2_l[l].b, h);
        mask_mul<<<eg(tot), 256, 0, stream>>>(h, maskp, tot);
    }

    // ---- slot pooling
    gemm_lin<0, 4><<<dim3(BNq / 16, 2, 1), 32, 0, stream>>>(
        h, 0, w_key, 0, pix_key.b, t128, 0, 128, 128, 1.f, nullptr, 0, 0);
    gemm_lin<0, 1><<<dim3(BNq / 16, 1, 1), 32, 0, stream>>>(
        t128, 0, w_q, 0, nullptr, pm, 0, 128, 16, 0.08838834764831845f, nullptr, 0, 0);
    zero_f<<<1, 128, 0, stream>>>(sums, 128);
    softmax_pool<<<eg(BNq), 256, 0, stream>>>(pm, maskp, pm, pmT, sums, Nn_);

    gemm_lin<0, 4><<<dim3(BNq / 16, 2, 1), 32, 0, stream>>>(
        h, 0, w_val, 0, pix_val.b, t128, 0, 128, 128, 1.f, nullptr, 0, 0);
    prep_wt<<<dim3(eg((long long)Nn_ * 128), 1, Bq), 256, 0, stream>>>(
        t128, vT16, Nn_, 128, (long long)Nn_ * 128, (long long)128 * Nn_);
    gemm_lin<0, 4><<<dim3(1, 2, Bq), 32, 0, stream>>>(
        pmT, (long long)Kq * Nn_, vT16, (long long)HDq * Nn_, nullptr,
        pf, (long long)Kq * HDq, Nn_, 128, 1.f, sums, Kq, 0);

    centers_k<<<128, 32, 0, stream>>>(pmT, sums, cen, Nn_);
    pos1_k<<<128, 128, 0, stream>>>(cen, pos1.w, pos1.b, pt);
    gemm_lin<0, 4><<<dim3(8, 2, 1), 32, 0, stream>>>(
        pt, 0, w_pos2, 0, pos2.b, pf, 0, 128, 128, 1.f, nullptr, 0, /*acc=*/1);

    // ---- part transformer layer
    gemm_lin<0, 4><<<dim3(8, 6, 1), 32, 0, stream>>>(
        pf, 0, w_pin, 0, p_inp.b, pqkv, 0, 128, 384, 1.f, nullptr, 0, 0);
    part_attn<<<Bq * 4, 32, 0, stream>>>(pqkv, ao);
    gemm_lin<0, 4><<<dim3(8, 2, 1), 32, 0, stream>>>(
        ao, 0, w_pout, 0, p_out.b, pt, 0, 128, 128, 1.f, nullptr, 0, 0);
    ln_res<<<128, 32, 0, stream>>>(pf, pt, p_ln1.g, p_ln1.b, pf);
    gemm_lin<1, 4><<<dim3(8, 4, 1), 32, 0, stream>>>(
        pf, 0, w_pffn1, 0, p_ffn1.b, p256, 0, 128, 256, 1.f, nullptr, 0, 0);
    gemm_lin<0, 4><<<dim3(8, 2, 1), 32, 0, stream>>>(
        p256, 0, w_pffn2, 0, p_ffn2.b, pt, 0, 256, 128, 1.f, nullptr, 0, 0);
    ln_res<<<128, 32, 0, stream>>>(pf, pt, p_ln2.g, p_ln2.b, pf);

    // ---- classifier head
    head_k<<<Bq, 128, 0, stream>>>(pf, cls1.w, cls1.b, cls2.w, cls2.b, (float*)d_out);
}
